// NetGINE_79285096284186
// MI455X (gfx1250) — matrix operand order repaired and checked by reading, compiled, not run on hardware
//
#include <hip/hip_runtime.h>
#include <hip/hip_bf16.h>

#define DIM 128
#define NN  50000
#define NE  1600000
#define NSUB 512
#define NGRAPH 64

typedef __attribute__((ext_vector_type(16))) _Float16 v16h;
typedef __attribute__((ext_vector_type(8)))  _Float16 v8h;
typedef __attribute__((ext_vector_type(4)))  _Float16 v4h;
typedef __attribute__((ext_vector_type(8)))  float    v8f;

union V16H { v16h v; v8h h[2]; };

// native f32 atomic add (global_atomic_add_f32)
__device__ __forceinline__ void atomAddF(float* p, float v) {
    unsafeAtomicAdd(p, v);
}

// ---- WMMA fragment loaders (16x16x32 f16, wave32) -------------------------
__device__ __forceinline__ v16h load_a_frag(const _Float16* tile, int kt) {
    const int lane = threadIdx.x & 31;
    const int m = lane & 15;
    const int g = lane >> 4;
    const _Float16* p = tile + m * DIM + kt * 32 + g * 8;
    V16H r;
    r.h[0] = *(const v8h*)(p);
    r.h[1] = *(const v8h*)(p + 16);
    return r.v;
}

__device__ __forceinline__ v16h load_b_frag(const _Float16* W, int kt, int nt) {
    const int lane = threadIdx.x & 31;
    const int n = lane & 15;
    const int g = lane >> 4;
    v16h r;
#pragma unroll
    for (int s = 0; s < 16; ++s) {
        int k = (s < 8) ? (g * 8 + s) : (16 + g * 8 + (s - 8));
        r[s] = W[(kt * 32 + k) * DIM + nt * 16 + n];
    }
    return r;
}

__device__ __forceinline__ v8f wmma_f16(v16h a, v16h b, v8f c) {
    return __builtin_amdgcn_wmma_f32_16x16x32_f16(false, a, false, b,
                                                  (short)0, c, false, false);
}

// ---------------------------------------------------------------------------
__global__ void k_zero(float4* __restrict__ p, long n4) {
    long i = (long)blockIdx.x * blockDim.x + threadIdx.x;
    long st = (long)gridDim.x * blockDim.x;
    float4 z = {0.f, 0.f, 0.f, 0.f};
    for (; i < n4; i += st) p[i] = z;
}

// ---------------------------------------------------------------------------
// conv1 edge path (28-dim, scalar VALU): agg1[dst] += relu(x[src] + bond1(ea))
__global__ __launch_bounds__(256) void k_edge_conv1(
    const float* __restrict__ x, const int* __restrict__ ei,
    const float* __restrict__ ea, const float* __restrict__ wba,
    const float* __restrict__ wbb, float* __restrict__ agg1)
{
    __shared__ float sWa[3 * 28];
    __shared__ float sWb[28 * 28];
    int tid = threadIdx.x;
    for (int i = tid; i < 3 * 28; i += 256) sWa[i] = wba[i];
    for (int i = tid; i < 28 * 28; i += 256) sWb[i] = wbb[i];
    __syncthreads();
    int e = blockIdx.x * 256 + tid;   // NE % 256 == 0: no tail
    float a0 = ea[e * 3], a1 = ea[e * 3 + 1], a2 = ea[e * 3 + 2];
    float t[28];
#pragma unroll
    for (int k = 0; k < 28; ++k) {
        float h = fmaf(a0, sWa[k], fmaf(a1, sWa[28 + k], a2 * sWa[56 + k]));
        t[k] = h > 0.f ? h : 0.f;
    }
    int so = ei[e] * 28, dof = ei[NE + e] * 28;
#pragma unroll
    for (int c = 0; c < 28; ++c) {
        float ec = 0.f;
#pragma unroll
        for (int k = 0; k < 28; ++k) ec = fmaf(t[k], sWb[k * 28 + c], ec);
        float v = x[so + c] + ec;
        atomAddF(&agg1[dof + c], v > 0.f ? v : 0.f);
    }
}

// conv1 node MLP: x1 = relu(relu(((1+eps)x + agg1) @ Wa) @ Wb)   28->28->128
__global__ __launch_bounds__(128) void k_node_mlp1(
    const float* __restrict__ x, const float* __restrict__ agg1,
    const float* __restrict__ wma, const float* __restrict__ wmb,
    const float* __restrict__ epsp, float* __restrict__ x1)
{
    __shared__ float sA[28 * 28];
    __shared__ float sB[28 * DIM];
    __shared__ float sh[28], st[28];
    int tid = threadIdx.x;
    for (int i = tid; i < 28 * 28; i += 128) sA[i] = wma[i];
    for (int i = tid; i < 28 * DIM; i += 128) sB[i] = wmb[i];
    float ep = 1.f + *epsp;
    __syncthreads();
    for (int n = blockIdx.x; n < NN; n += gridDim.x) {
        if (tid < 28) sh[tid] = ep * x[n * 28 + tid] + agg1[n * 28 + tid];
        __syncthreads();
        if (tid < 28) {
            float a = 0.f;
            for (int k = 0; k < 28; ++k) a = fmaf(sh[k], sA[k * 28 + tid], a);
            st[tid] = a > 0.f ? a : 0.f;
        }
        __syncthreads();
        float a = 0.f;
#pragma unroll
        for (int k = 0; k < 28; ++k) a = fmaf(st[k], sB[k * DIM + tid], a);
        x1[n * DIM + tid] = a > 0.f ? a : 0.f;
        __syncthreads();
    }
}

// ---------------------------------------------------------------------------
// conv2 edge path. Block: 8 waves, 4 edge tiles of 128 edges (512 edges/blk).
// NE % 512 == 0 -> no tail guards. Single accumulator chain per N-tile keeps
// VGPRs under the 256 window (occupancy > ILP for this latency-bound kernel).
#define ETILES 4
__global__ __launch_bounds__(256) void k_edge_conv2(
    const float* __restrict__ x1, const int* __restrict__ ei,
    const float* __restrict__ ea, const float* __restrict__ wba,
    const float* __restrict__ wbb, float* __restrict__ agg2)
{
    __shared__ __align__(32) _Float16 sW[DIM * DIM];    // Wbb f16, 32KB
    __shared__ __align__(32) _Float16 sH[8][16 * DIM];  // per-wave hidden tile
    __shared__ float sWa[3 * DIM];

    const int tid = threadIdx.x;
    for (int i = tid; i < DIM * DIM; i += 256) sW[i] = (_Float16)wbb[i];
    for (int i = tid; i < 3 * DIM; i += 256) sWa[i] = wba[i];
    __syncthreads();

    const int wave = tid >> 5;
    const int lane = tid & 31;
    const int m = lane & 15;
    const int g = lane >> 4;
    const int n16 = lane & 15;
    _Float16* myH = sH[wave];

#pragma unroll 1
    for (int t = 0; t < ETILES; ++t) {
        const int e0 = (blockIdx.x * ETILES + t) * 128 + wave * 16;

        // stage 1: hidden = relu(ea @ Wba); packed v8h stores (ds_store_b128)
        const int em = e0 + m;
        float a0 = ea[em * 3 + 0];
        float a1 = ea[em * 3 + 1];
        float a2 = ea[em * 3 + 2];
#pragma unroll
        for (int c8 = 0; c8 < 8; ++c8) {
            v8h pk;
#pragma unroll
            for (int u = 0; u < 8; ++u) {
                int col = g * 64 + c8 * 8 + u;
                float h = fmaf(a0, sWa[col], fmaf(a1, sWa[DIM + col], a2 * sWa[2 * DIM + col]));
                pk[u] = (_Float16)(h > 0.f ? h : 0.f);
            }
            *(v8h*)(myH + m * DIM + g * 64 + c8 * 8) = pk;
        }
        // wave-local LDS RAW: DS ops from one wave complete in order

        v16h af[4];
#pragma unroll
        for (int kt = 0; kt < 4; ++kt) af[kt] = load_a_frag(myH, kt);

        int so[8], dof[8];                    // 32-bit element offsets
#pragma unroll
        for (int r = 0; r < 8; ++r) {
            int e = e0 + r + g * 8;           // rows this lane owns in C
            so[r] = ei[e] * DIM;
            dof[r] = ei[NE + e] * DIM;
        }

#pragma unroll 1
        for (int nt = 0; nt < 8; ++nt) {
            v16h bf[4];
#pragma unroll
            for (int kt = 0; kt < 4; ++kt) bf[kt] = load_b_frag(sW, kt, nt);
            v8f acc = {};
#pragma unroll
            for (int kt = 0; kt < 4; ++kt) acc = wmma_f16(af[kt], bf[kt], acc);

            const int col = nt * 16 + n16;
            float gv[8];
#pragma unroll
            for (int r = 0; r < 8; ++r) gv[r] = x1[so[r] + col];  // batched gather
#pragma unroll
            for (int r = 0; r < 8; ++r) {
                float v = gv[r] + acc[r];
                v = v > 0.f ? v : 0.f;
                atomAddF(&agg2[dof[r] + col], v);
            }
        }
    }
}

// conv2 node MLP (WMMA): x2 = relu(relu(((1+eps)x1 + agg2) @ Wa) @ Wb)
__global__ __launch_bounds__(256) void k_node_mlp2(
    const float* __restrict__ x1, const float* __restrict__ agg2,
    const float* __restrict__ wma, const float* __restrict__ wmb,
    const float* __restrict__ epsp, float* __restrict__ x2)
{
    __shared__ __align__(32) _Float16 sW1[DIM * DIM];
    __shared__ __align__(32) _Float16 sW2[DIM * DIM];
    __shared__ __align__(32) _Float16 sA[128 * DIM];  // node tile, reused for T

    const int tid = threadIdx.x;
    for (int i = tid; i < DIM * DIM; i += 256) {
        sW1[i] = (_Float16)wma[i];
        sW2[i] = (_Float16)wmb[i];
    }
    const float ep = 1.f + *epsp;
    const int base = blockIdx.x * 128;
    const bool full = (base + 128 <= NN);     // uniform: only last block partial

    const float4* x1v = (const float4*)x1;
    const float4* agv = (const float4*)agg2;
    for (int i = tid; i < 128 * DIM / 4; i += 256) {  // float4 loads, v4h stores
        int node = base + (i >> 5);
        int c4 = i & 31;
        v4h pk = {};
        if (full || node < NN) {
            float4 xa = x1v[node * 32 + c4];
            float4 ag = agv[node * 32 + c4];
            pk[0] = (_Float16)fmaf(ep, xa.x, ag.x);
            pk[1] = (_Float16)fmaf(ep, xa.y, ag.y);
            pk[2] = (_Float16)fmaf(ep, xa.z, ag.z);
            pk[3] = (_Float16)fmaf(ep, xa.w, ag.w);
        }
        *(v4h*)(sA + i * 4) = pk;
    }
    __syncthreads();

    const int wave = tid >> 5;
    const int lane = tid & 31;
    const int g = lane >> 4;
    const int n16 = lane & 15;
    _Float16* myA = sA + wave * 16 * DIM;

    v16h af[4];
#pragma unroll
    for (int kt = 0; kt < 4; ++kt) af[kt] = load_a_frag(myA, kt);

    // GEMM1: T = relu(hin @ Wa), stored back over this wave's own tile
#pragma unroll 1
    for (int nt = 0; nt < 8; ++nt) {
        v16h bf[4];
#pragma unroll
        for (int kt = 0; kt < 4; ++kt) bf[kt] = load_b_frag(sW1, kt, nt);
        v8f acc = {};
#pragma unroll
        for (int kt = 0; kt < 4; ++kt) acc = wmma_f16(af[kt], bf[kt], acc);
#pragma unroll
        for (int r = 0; r < 8; ++r) {
            float v = acc[r];
            myA[(r + g * 8) * DIM + nt * 16 + n16] = (_Float16)(v > 0.f ? v : 0.f);
        }
    }
    __syncthreads();

    // GEMM2: x2 = relu(T @ Wb)
#pragma unroll
    for (int kt = 0; kt < 4; ++kt) af[kt] = load_a_frag(myA, kt);
#pragma unroll 1
    for (int nt = 0; nt < 8; ++nt) {
        v16h bf[4];
#pragma unroll
        for (int kt = 0; kt < 4; ++kt) bf[kt] = load_b_frag(sW2, kt, nt);
        v8f acc = {};
#pragma unroll
        for (int kt = 0; kt < 4; ++kt) acc = wmma_f16(af[kt], bf[kt], acc);
        const int col = nt * 16 + n16;
#pragma unroll
        for (int r = 0; r < 8; ++r) {
            int node = base + wave * 16 + r + g * 8;
            float v = acc[r];
            if (full || node < NN)
                x2[node * DIM + col] = v > 0.f ? v : 0.f;
        }
    }
}

// ---------------------------------------------------------------------------
// mean-pool nodes -> subgraphs. batch is SORTED: accumulate per segment in
// registers, flush atomics only on segment change (~64x fewer atomics).
#define PNODES 128
__global__ __launch_bounds__(256) void k_pool_nodes(
    const float* __restrict__ x1, const float* __restrict__ x2,
    const int* __restrict__ batch, float* __restrict__ psum, float* __restrict__ pcnt)
{
    int n0 = blockIdx.x * PNODES;
    int c = threadIdx.x;
    const float* srcp = (c < DIM) ? (x1 + c) : (x2 + (c - DIM));
    float acc = 0.f;
    int cnt = 0;
    int curb = batch[n0];
    for (int i = 0; i < PNODES; ++i) {
        int n = n0 + i;
        if (n >= NN) break;
        int b = batch[n];                 // uniform scalar load
        if (b != curb) {
            atomAddF(&psum[curb * 256 + c], acc);
            if (c == 0) atomAddF(&pcnt[curb], (float)cnt);
            acc = 0.f; cnt = 0; curb = b;
        }
        acc += srcp[n * DIM];
        ++cnt;
    }
    atomAddF(&psum[curb * 256 + c], acc);
    if (c == 0) atomAddF(&pcnt[curb], (float)cnt);
}

// fc1 + relu + fc2 + relu per subgraph, scatter into graph sums
__global__ __launch_bounds__(128) void k_fc12(
    const float* __restrict__ psum, const float* __restrict__ pcnt,
    const int* __restrict__ gidx,
    const float* __restrict__ w1, const float* __restrict__ b1,
    const float* __restrict__ w2, const float* __restrict__ b2,
    float* __restrict__ gsum, float* __restrict__ gcnt)
{
    __shared__ float sp[256];
    __shared__ float sh[128];
    int s = blockIdx.x;
    int j = threadIdx.x;
    float cnt = pcnt[s]; cnt = cnt > 1.f ? cnt : 1.f;
    sp[j]       = psum[s * 256 + j] / cnt;
    sp[j + 128] = psum[s * 256 + 128 + j] / cnt;
    __syncthreads();
    float a = b1[j];
    for (int k = 0; k < 256; ++k) a = fmaf(sp[k], w1[k * 128 + j], a);
    sh[j] = a > 0.f ? a : 0.f;
    __syncthreads();
    a = b2[j];
    for (int k = 0; k < 128; ++k) a = fmaf(sh[k], w2[k * 128 + j], a);
    a = a > 0.f ? a : 0.f;
    int gg = gidx[s];
    atomAddF(&gsum[gg * 128 + j], a);
    if (j == 0) atomAddF(&gcnt[gg], 1.f);
}

// graph mean + fc3 -> out[64]
__global__ __launch_bounds__(128) void k_fc3(
    const float* __restrict__ gsum, const float* __restrict__ gcnt,
    const float* __restrict__ w3, const float* __restrict__ b3,
    float* __restrict__ out)
{
    __shared__ float red[128];
    int gph = blockIdx.x, j = threadIdx.x;
    float cnt = gcnt[gph]; cnt = cnt > 1.f ? cnt : 1.f;
    red[j] = (gsum[gph * 128 + j] / cnt) * w3[j];
    __syncthreads();
    for (int st = 64; st > 0; st >>= 1) {
        if (j < st) red[j] += red[j + st];
        __syncthreads();
    }
    if (j == 0) out[gph] = red[0] + b3[0];
}

// ---------------------------------------------------------------------------
extern "C" void kernel_launch(void* const* d_in, const int* in_sizes, int n_in,
                              void* d_out, int out_size, void* d_ws, size_t ws_size,
                              hipStream_t stream) {
    const float* x     = (const float*)d_in[0];
    const int*   ei    = (const int*)d_in[1];
    const float* ea    = (const float*)d_in[2];
    const int*   batch = (const int*)d_in[3];
    const int*   gidx  = (const int*)d_in[4];
    const float* w_b1a = (const float*)d_in[5];
    const float* w_b1b = (const float*)d_in[6];
    const float* w_m1a = (const float*)d_in[7];
    const float* w_m1b = (const float*)d_in[8];
    const float* eps1  = (const float*)d_in[9];
    const float* w_b2a = (const float*)d_in[10];
    const float* w_b2b = (const float*)d_in[11];
    const float* w_m2a = (const float*)d_in[12];
    const float* w_m2b = (const float*)d_in[13];
    const float* eps2  = (const float*)d_in[14];
    const float* w_fc1 = (const float*)d_in[15];
    const float* b_fc1 = (const float*)d_in[16];
    const float* w_fc2 = (const float*)d_in[17];
    const float* b_fc2 = (const float*)d_in[18];
    const float* w_fc3 = (const float*)d_in[19];
    const float* b_fc3 = (const float*)d_in[20];
    float* out = (float*)d_out;

    // workspace layout: zero-initialized region first
    float* ws = (float*)d_ws;
    size_t o = 0;
    float* agg1 = ws + o; o += (size_t)NN * 28;
    float* agg2 = ws + o; o += (size_t)NN * DIM;
    float* psum = ws + o; o += (size_t)NSUB * 256;
    float* pcnt = ws + o; o += NSUB;
    float* gsum = ws + o; o += (size_t)NGRAPH * 128;
    float* gcnt = ws + o; o += NGRAPH;
    size_t zn = o;
    float* x1 = ws + o; o += (size_t)NN * DIM;
    float* x2 = ws + o; o += (size_t)NN * DIM;

    k_zero<<<2048, 256, 0, stream>>>((float4*)ws, (long)(zn / 4));
    k_edge_conv1<<<NE / 256, 256, 0, stream>>>(x, ei, ea, w_b1a, w_b1b, agg1);
    k_node_mlp1<<<2048, 128, 0, stream>>>(x, agg1, w_m1a, w_m1b, eps1, x1);
    k_edge_conv2<<<NE / (128 * ETILES), 256, 0, stream>>>(x1, ei, ea, w_b2a, w_b2b, agg2);
    k_node_mlp2<<<(NN + 127) / 128, 256, 0, stream>>>(x1, agg2, w_m2a, w_m2b, eps2, x2);
    k_pool_nodes<<<(NN + PNODES - 1) / PNODES, 256, 0, stream>>>(x1, x2, batch, psum, pcnt);
    k_fc12<<<NSUB, 128, 0, stream>>>(psum, pcnt, gidx, w_fc1, b_fc1, w_fc2, b_fc2, gsum, gcnt);
    k_fc3<<<NGRAPH, 128, 0, stream>>>(gsum, gcnt, w_fc3, b_fc3, out);
}
